// DGCNN_47004122088066
// MI455X (gfx1250) — matrix operand order repaired and checked by reading, not compiled, not run on hardware
//
#include <hip/hip_runtime.h>
#include <hip/hip_bf16.h>
#include <math.h>

// ---------------- static problem dimensions ----------------
#define NUM_GRAPHS 64
#define NODES_PER_GRAPH 512
#define N_TOTAL (NUM_GRAPHS * NODES_PER_GRAPH)   // 32768
#define E_TOTAL (N_TOTAL * 16)                   // 524288
#define EMB 128
#define DIMF 128
#define NUM_LAYERS 4
#define KTOP 64
#define LATENT (NUM_LAYERS * DIMF + 1)           // 513

typedef __attribute__((ext_vector_type(16))) __bf16 v16bf;
typedef __attribute__((ext_vector_type(8)))  __bf16 v8bf;
typedef __attribute__((ext_vector_type(8)))  float  v8f;

// ---------------- degree ----------------
__global__ void k_init_deg(float* __restrict__ deg) {
    int i = blockIdx.x * blockDim.x + threadIdx.x;
    if (i < N_TOTAL) deg[i] = 1.0f;   // self loop
}

__global__ void k_edge_deg(const int* __restrict__ src, float* __restrict__ deg) {
    int e = blockIdx.x * blockDim.x + threadIdx.x;
    if (e < E_TOTAL) atomicAdd(&deg[src[e]], 1.0f);
}

// ---------------- weight pre-convert: W[l][k][n] f32 -> Wt[l][n][k] bf16 ----------------
__global__ void k_convert_Wt(const float* __restrict__ W, __bf16* __restrict__ Wt) {
    int idx = blockIdx.x * blockDim.x + threadIdx.x;     // over 4*128*128
    int l = idx >> 14, r = idx & 16383;
    int n = r >> 7, k = r & 127;
    Wt[idx] = (__bf16)W[l * 16384 + k * 128 + n];
}

// ---------------- embedding gather (f32 + bf16 staging) ----------------
__global__ void k_gather_emb(const int* __restrict__ x, const float* __restrict__ emb,
                             float* __restrict__ h0, __bf16* __restrict__ h0bf) {
    long idx = (long)blockIdx.x * blockDim.x + threadIdx.x;   // over N*128
    int i = (int)(idx >> 7), f = (int)(idx & 127);
    float v = emb[(long)x[i] * EMB + f];
    h0[idx] = v;
    h0bf[idx] = (__bf16)v;
}

// ---------------- GEMM: T = H(Mx128) @ W(128x128) + bias, bf16 WMMA ----------------
// A: bf16 row-major [M][128]; B: bf16 col-major Wt[n][k]. One wave per 16x16 tile.
__global__ __launch_bounds__(256) void k_gemm_wmma(const __bf16* __restrict__ Hbf,
                                                   const __bf16* __restrict__ Wt,
                                                   const float* __restrict__ bias,
                                                   float* __restrict__ T) {
    int wave = blockIdx.x * (blockDim.x >> 5) + (threadIdx.x >> 5);
    int lane = threadIdx.x & 31;
    int tm = wave >> 3;        // M tile
    int tn = wave & 7;         // N tile
    int r15 = lane & 15;
    int kh  = lane >> 4;       // K-half selector (this code's contiguous k = 16*kh + j order)

    int rowA = tm * 16 + r15;
    int colB = tn * 16 + r15;
    const __bf16* arow = Hbf + (long)rowA * 128;
    const __bf16* bcol = Wt + (long)colB * 128;
    v8f c = {};

#pragma unroll
    for (int kk = 0; kk < 128; kk += 32) {
        // A fragment: a[j] = H[rowA][K = kk + kh*16 + j]
        v8bf a0 = *(const v8bf*)(arow + kk + kh * 16);
        v8bf a1 = *(const v8bf*)(arow + kk + kh * 16 + 8);
        v16bf a = __builtin_shufflevector(a0, a1, 0, 1, 2, 3, 4, 5, 6, 7,
                                          8, 9, 10, 11, 12, 13, 14, 15);
        // B fragment: b[j] = W[K = kk + kh*16 + j][colB] = Wt[colB][kk + kh*16 + j]
        v8bf b0 = *(const v8bf*)(bcol + kk + kh * 16);
        v8bf b1 = *(const v8bf*)(bcol + kk + kh * 16 + 8);
        v16bf b = __builtin_shufflevector(b0, b1, 0, 1, 2, 3, 4, 5, 6, 7,
                                          8, 9, 10, 11, 12, 13, 14, 15);

        c = __builtin_amdgcn_wmma_f32_16x16x32_bf16(
                false, a, false, b, (short)0, c, false, false);
    }

    // C/D: VGPR r -> row r + 8*kh, col = lane&15
    float bi = bias[colB];
#pragma unroll
    for (int r = 0; r < 8; ++r) {
        int rowg = tm * 16 + r + 8 * kh;
        T[(long)rowg * 128 + colB] = c[r] + bi;
    }
}

// ---------------- edge scatter: OUT[dst] += T[src] ----------------
__global__ void k_scatter_vec(const int* __restrict__ src, const int* __restrict__ dst,
                              const float* __restrict__ T, float* __restrict__ out) {
    long idx = (long)blockIdx.x * blockDim.x + threadIdx.x;   // over E*128
    int e = (int)(idx >> 7), f = (int)(idx & 127);
    atomicAdd(&out[(long)dst[e] * 128 + f], T[(long)src[e] * 128 + f]);
}

// ---------------- finalize: h = tanh(acc / deg), also stage bf16 for next GEMM ----------
__global__ void k_finalize_vec(float* __restrict__ h, const float* __restrict__ deg,
                               __bf16* __restrict__ hbf) {
    long idx = (long)blockIdx.x * blockDim.x + threadIdx.x;   // over N*128
    int i = (int)(idx >> 7);
    float v = tanhf(h[idx] / deg[i]);
    h[idx] = v;
    hbf[idx] = (__bf16)v;
}

// ---------------- last layer: per-node dot with W_last[128] ----------------
__global__ void k_last_dot(const float* __restrict__ h, const float* __restrict__ Wl,
                           const float* __restrict__ bl, float* __restrict__ s) {
    int i = blockIdx.x * blockDim.x + threadIdx.x;
    if (i >= N_TOTAL) return;
    float acc = bl[0];
    const float* hp = h + (long)i * 128;
#pragma unroll 8
    for (int f = 0; f < 128; ++f) acc += hp[f] * Wl[f];
    s[i] = acc;
}

__global__ void k_scatter_scalar(const int* __restrict__ src, const int* __restrict__ dst,
                                 const float* __restrict__ s, float* __restrict__ out) {
    int e = blockIdx.x * blockDim.x + threadIdx.x;
    if (e < E_TOTAL) atomicAdd(&out[dst[e]], s[src[e]]);
}

__global__ void k_finalize_scalar(float* __restrict__ h, const float* __restrict__ deg) {
    int i = blockIdx.x * blockDim.x + threadIdx.x;
    if (i < N_TOTAL) h[i] = tanhf(h[i] / deg[i]);
}

// ---------------- global_sort_pool: per-graph bitonic top-K ----------------
// stable argsort(-key): descending key, ties -> smaller original index first
__global__ __launch_bounds__(512) void k_sortpool(const float* __restrict__ hlast,
                                                  int* __restrict__ order) {
    __shared__ float key[NODES_PER_GRAPH];
    __shared__ int   idx[NODES_PER_GRAPH];
    int g = blockIdx.x, t = threadIdx.x;
    key[t] = hlast[g * NODES_PER_GRAPH + t];
    idx[t] = t;
    for (int k = 2; k <= NODES_PER_GRAPH; k <<= 1) {
        for (int j = k >> 1; j > 0; j >>= 1) {
            __syncthreads();
            int ixj = t ^ j;
            if (ixj > t) {
                float ka = key[t], kb = key[ixj];
                int   ia = idx[t], ib = idx[ixj];
                bool aBeforeB = (ka > kb) || (ka == kb && ia < ib);
                bool dirUp = ((t & k) == 0);
                if (dirUp ? !aBeforeB : aBeforeB) {
                    key[t] = kb; key[ixj] = ka;
                    idx[t] = ib; idx[ixj] = ia;
                }
            }
        }
    }
    __syncthreads();
    if (t < KTOP) order[g * KTOP + t] = idx[t];
}

// ---------------- conv1: per (graph,k) 16 projections over 513 features ----------------
__global__ void k_conv1(const float* __restrict__ hid,    // [4][N][128]
                        const float* __restrict__ hlast,  // [N]
                        const int*   __restrict__ order,  // [B][K]
                        const float* __restrict__ w1,     // [16][513]
                        const float* __restrict__ b1,     // [16]
                        float* __restrict__ y1) {         // [B][16][K]
    __shared__ float vec[LATENT];
    int bid = blockIdx.x;
    int b = bid >> 6, k = bid & 63;
    int node = b * NODES_PER_GRAPH + order[b * KTOP + k];
    for (int d = threadIdx.x; d < LATENT; d += blockDim.x) {
        if (d < NUM_LAYERS * DIMF) {
            int l = d >> 7, f = d & 127;
            vec[d] = hid[((long)l * N_TOTAL + node) * 128 + f];
        } else {
            vec[d] = hlast[node];
        }
    }
    __syncthreads();
    if (threadIdx.x < 16) {
        int o = threadIdx.x;
        float acc = b1[o];
        const float* wp = w1 + (long)o * LATENT;
        for (int d = 0; d < LATENT; ++d) acc += vec[d] * wp[d];
        y1[((long)b * 16 + o) * KTOP + k] = fmaxf(acc, 0.0f);
    }
}

// ---------------- head: maxpool2 + conv2(k=5) + dense(896->32) + dense(32->2) + softmax ----
__global__ __launch_bounds__(256) void k_head(const float* __restrict__ y1,    // [B][16][64]
                       const float* __restrict__ w2, const float* __restrict__ b2,   // [32][16][5],[32]
                       const float* __restrict__ d1w, const float* __restrict__ d1b, // [896][32],[32]
                       const float* __restrict__ d2w, const float* __restrict__ d2b, // [32][2],[2]
                       float* __restrict__ out) {                                    // [B][2]
    __shared__ float y1l[16 * 64];
    __shared__ float p[16 * 32];
    __shared__ float y2f[896];
    __shared__ float h1[32];
    __shared__ float logits[2];
    int b = blockIdx.x, t = threadIdx.x;

    for (int e = t; e < 16 * 64; e += blockDim.x)
        y1l[e] = y1[(long)b * 16 * 64 + e];
    __syncthreads();

    for (int e = t; e < 16 * 32; e += blockDim.x) {
        int i = e >> 5, pos = e & 31;
        p[e] = fmaxf(y1l[i * 64 + 2 * pos], y1l[i * 64 + 2 * pos + 1]);
    }
    __syncthreads();

    for (int e = t; e < 32 * 28; e += blockDim.x) {
        int o = e / 28, pos = e % 28;
        float acc = b2[o];
#pragma unroll
        for (int i = 0; i < 16; ++i)
#pragma unroll
            for (int kk = 0; kk < 5; ++kk)
                acc += p[i * 32 + pos + kk] * w2[(o * 16 + i) * 5 + kk];
        y2f[o * 28 + pos] = fmaxf(acc, 0.0f);
    }
    __syncthreads();

    if (t < 32) {
        float acc = d1b[t];
        for (int m = 0; m < 896; ++m) acc += y2f[m] * d1w[m * 32 + t];
        h1[t] = fmaxf(acc, 0.0f);
    }
    __syncthreads();

    if (t < 2) {
        float acc = d2b[t];
#pragma unroll
        for (int j = 0; j < 32; ++j) acc += h1[j] * d2w[j * 2 + t];
        logits[t] = acc;
    }
    __syncthreads();

    if (t == 0) {
        float m = fmaxf(logits[0], logits[1]);
        float e0 = expf(logits[0] - m), e1 = expf(logits[1] - m);
        float s = e0 + e1;
        out[b * 2 + 0] = e0 / s;
        out[b * 2 + 1] = e1 / s;
    }
}

// ---------------- launcher ----------------
extern "C" void kernel_launch(void* const* d_in, const int* in_sizes, int n_in,
                              void* d_out, int out_size, void* d_ws, size_t ws_size,
                              hipStream_t stream) {
    const int*   x       = (const int*)d_in[0];
    const int*   src     = (const int*)d_in[1];             // edge_index[0]
    const int*   dst     = ((const int*)d_in[1]) + E_TOTAL; // edge_index[1]
    const float* emb     = (const float*)d_in[3];
    const float* W_convs = (const float*)d_in[4];
    const float* b_convs = (const float*)d_in[5];
    const float* W_last  = (const float*)d_in[6];
    const float* b_last  = (const float*)d_in[7];
    const float* conv1_w = (const float*)d_in[8];
    const float* conv1_b = (const float*)d_in[9];
    const float* conv2_w = (const float*)d_in[10];
    const float* conv2_b = (const float*)d_in[11];
    const float* d1_w    = (const float*)d_in[12];
    const float* d1_b    = (const float*)d_in[13];
    const float* d2_w    = (const float*)d_in[14];
    const float* d2_b    = (const float*)d_in[15];
    float* out = (float*)d_out;

    // workspace carve-up
    float* ws   = (float*)d_ws;
    const long NF = (long)N_TOTAL * 128;
    float* hid   = ws;                         // [4][N][128] f32
    float* T     = hid + 4 * NF;               // [N][128] f32
    float* deg   = T + NF;                     // [N]
    float* slast = deg + N_TOTAL;              // [N]
    float* hlast = slast + N_TOTAL;            // [N]
    float* y1    = hlast + N_TOTAL;            // [B][16][64]
    int*   order = (int*)(y1 + NUM_GRAPHS * 16 * KTOP);     // [B][K]
    __bf16* Wbf  = (__bf16*)(order + NUM_GRAPHS * KTOP);    // [4][128][128] (col-major per layer)
    __bf16* h0bf = Wbf + 4 * 128 * 128;        // [N][128] bf16
    __bf16* hbf  = h0bf + NF;                  // [N][128] bf16 staging

    // degree (self loop + edge src counts)
    k_init_deg<<<(N_TOTAL + 255) / 256, 256, 0, stream>>>(deg);
    k_edge_deg<<<(E_TOTAL + 255) / 256, 256, 0, stream>>>(src, deg);

    // weight pre-conversion (f32 k-major -> bf16 n-major) and embedding gather
    k_convert_Wt<<<(4 * 128 * 128) / 256, 256, 0, stream>>>(W_convs, Wbf);
    k_gather_emb<<<(int)(NF / 256), 256, 0, stream>>>(x, emb, hid /*unused slot ok*/, h0bf);
    // note: layer-0 GEMM consumes only the bf16 copy; hid[0] is overwritten below.

    // GCN layers
    const int gemm_blocks = (N_TOTAL / 16) * 8 / 8;   // tiles / wavesPerBlock = 2048
    for (int l = 0; l < NUM_LAYERS; ++l) {
        const __bf16* Hin = (l == 0) ? h0bf : hbf;
        float* Hout = hid + (long)l * NF;
        k_gemm_wmma<<<gemm_blocks, 256, 0, stream>>>(
            Hin, Wbf + (long)l * 128 * 128, b_convs + l * 128, T);
        hipMemcpyAsync(Hout, T, NF * sizeof(float), hipMemcpyDeviceToDevice, stream);
        k_scatter_vec<<<(int)(((long)E_TOTAL * 128) / 256), 256, 0, stream>>>(src, dst, T, Hout);
        k_finalize_vec<<<(int)(NF / 256), 256, 0, stream>>>(Hout, deg, hbf);
    }

    // last projection + propagate
    k_last_dot<<<(N_TOTAL + 255) / 256, 256, 0, stream>>>(hid + 3 * NF, W_last, b_last, slast);
    hipMemcpyAsync(hlast, slast, N_TOTAL * sizeof(float), hipMemcpyDeviceToDevice, stream);
    k_scatter_scalar<<<(E_TOTAL + 255) / 256, 256, 0, stream>>>(src, dst, slast, hlast);
    k_finalize_scalar<<<(N_TOTAL + 255) / 256, 256, 0, stream>>>(hlast, deg);

    // sort pool + head
    k_sortpool<<<NUM_GRAPHS, NODES_PER_GRAPH, 0, stream>>>(hlast, order);
    k_conv1<<<NUM_GRAPHS * KTOP, 128, 0, stream>>>(hid, hlast, order, conv1_w, conv1_b, y1);
    k_head<<<NUM_GRAPHS, 256, 0, stream>>>(y1, conv2_w, conv2_b, d1_w, d1_b, d2_w, d2_b, out);
}
